// DynamicGCN_86397562126410
// MI455X (gfx1250) — matrix operand order repaired and checked by reading, not compiled, not run on hardware
//
#include <hip/hip_runtime.h>
#include <hip/hip_bf16.h>

typedef __attribute__((ext_vector_type(16))) _Float16 v16h;
typedef __attribute__((ext_vector_type(8)))  float    v8f;

#define BB 2
#define NN 512
#define DD 256
#define HH 256
#define H2 128

// ---------------------------------------------------------------------------
// Kernel 1: L'[b,i,h] = x[b,i,:]@W1[:D,h] + b1[h]   (b1 folded into L)
//           R [b,j,h] = x[b,j,:]@W1[D:,h]
// grid = B*N blocks, 256 threads (h = tid)
// ---------------------------------------------------------------------------
__global__ __launch_bounds__(256) void prep_lr(const float* __restrict__ x,
                                               const float* __restrict__ W1,
                                               const float* __restrict__ b1,
                                               float* __restrict__ Lp,
                                               float* __restrict__ Rp) {
    __shared__ float xs[DD];
    const int tid = threadIdx.x;
    const int row = blockIdx.x;                 // b*N + i
    xs[tid] = x[((size_t)row << 8) + tid];
    __syncthreads();
    float aL = b1[tid];
    float aR = 0.f;
    #pragma unroll 8
    for (int d = 0; d < DD; ++d) {
        const float xv = xs[d];
        aL += xv * W1[d * HH + tid];
        aR += xv * W1[(DD + d) * HH + tid];
    }
    Lp[((size_t)row << 8) + tid] = aL;
    Rp[((size_t)row << 8) + tid] = aR;
}

// ---------------------------------------------------------------------------
// Kernel 2: pack W2 [H=256, H2=128] fp32 -> f16 in per-lane WMMA B-operand
// layout: flat = (((kt*8 + ct)*32) + lane)*16 + p
//   B-matrix (32x16 f16, matches A layout): lanes 0-15 hold K=0-7,16-23, lanes 16-31
//   hold K=8-15,24-31; column n = ct*16 + (lane&15); K = kt*32 + 8*(lane>>4) + (p&7) + 16*(p>>3)
// ---------------------------------------------------------------------------
__global__ __launch_bounds__(256) void pack_w2(const float* __restrict__ W2,
                                               _Float16* __restrict__ W2p) {
    const int flat = blockIdx.x * 256 + threadIdx.x;   // 0..32767
    const int p  = flat & 15;
    const int l  = (flat >> 4) & 31;
    const int ct = (flat >> 9) & 7;
    const int kt = (flat >> 12) & 7;
    const int K  = kt * 32 + ((l >> 4) << 3) + (p & 7) + ((p >> 3) << 4);
    const int n  = (ct << 4) + (l & 15);
    W2p[flat] = (_Float16)W2[K * H2 + n];
}

// ---------------------------------------------------------------------------
// Kernel 3 (hot): fused edge MLP via WMMA.
// One wave = one tile of 16 pairs (fixed i, 16 consecutive j).
// A (16x32 f16) rebuilt in regs each k-step: relu(L_i + R_j); B from LDS.
// 8 k-steps x 8 col-tiles = 64 v_wmma_f32_16x16x32_f16 per wave.
// Epilogue: relu(+b2) . w3 -> xor-shuffle reduce -> sigmoid ->
//           E = base_adj*edge + I
// ---------------------------------------------------------------------------
__device__ __forceinline__ _Float16 relu_h(float a, float b) {
    const float s = a + b;
    return (_Float16)(s > 0.f ? s : 0.f);
}

__global__ __launch_bounds__(128) void edge_mlp(const float* __restrict__ Lp,
                                                const float* __restrict__ Rp,
                                                const _Float16* __restrict__ W2p,
                                                const float* __restrict__ b2,
                                                const float* __restrict__ W3,
                                                const float* __restrict__ b3,
                                                const float* __restrict__ base_adj,
                                                float* __restrict__ E) {
    __shared__ _Float16 w2s[32768];            // 64 KB packed W2
    const int tid = threadIdx.x;

    // cooperative 64KB stage (4096 x 16B, 128 threads -> 32 each)
    {
        const uint4* src = (const uint4*)W2p;
        uint4*       dst = (uint4*)w2s;
        for (int k = tid; k < 4096; k += 128) dst[k] = src[k];
    }
    __syncthreads();

    const int wave = tid >> 5;
    const int lane = tid & 31;
    const int tile = blockIdx.x * 4 + wave;    // 32768 tiles total
    const int jt   = tile & 31;                // N/16 = 32 j-tiles
    const int i    = (tile >> 5) & 511;
    const int b    = tile >> 14;               // 16384 tiles per batch
    const int jbase = jt << 4;

    const int m  = lane & 15;                  // pair row within tile (= j offset)
    const int kh = (lane >> 4) << 3;           // 0 or 8 (A-layout K half)

    const float* Lrow = Lp + ((size_t)(b * NN + i) << 8);
    const float* Rrow = Rp + ((size_t)(b * NN + jbase + m) << 8);

    v8f acc[8];
    #pragma unroll
    for (int ct = 0; ct < 8; ++ct) acc[ct] = (v8f){0.f,0.f,0.f,0.f,0.f,0.f,0.f,0.f};

    const v16h* w2v = (const v16h*)w2s;

    #pragma unroll
    for (int kt = 0; kt < 8; ++kt) {
        const int k0 = kt * 32;
        // A-layout runs: K = k0+kh+0..7 and K = k0+16+kh+0..7 (32B-aligned)
        const float4 l0 = *(const float4*)(Lrow + k0 + kh);
        const float4 l1 = *(const float4*)(Lrow + k0 + kh + 4);
        const float4 l2 = *(const float4*)(Lrow + k0 + 16 + kh);
        const float4 l3 = *(const float4*)(Lrow + k0 + 16 + kh + 4);
        const float4 r0 = *(const float4*)(Rrow + k0 + kh);
        const float4 r1 = *(const float4*)(Rrow + k0 + kh + 4);
        const float4 r2 = *(const float4*)(Rrow + k0 + 16 + kh);
        const float4 r3 = *(const float4*)(Rrow + k0 + 16 + kh + 4);

        v16h a;
        a[0]  = relu_h(l0.x, r0.x);  a[1]  = relu_h(l0.y, r0.y);
        a[2]  = relu_h(l0.z, r0.z);  a[3]  = relu_h(l0.w, r0.w);
        a[4]  = relu_h(l1.x, r1.x);  a[5]  = relu_h(l1.y, r1.y);
        a[6]  = relu_h(l1.z, r1.z);  a[7]  = relu_h(l1.w, r1.w);
        a[8]  = relu_h(l2.x, r2.x);  a[9]  = relu_h(l2.y, r2.y);
        a[10] = relu_h(l2.z, r2.z);  a[11] = relu_h(l2.w, r2.w);
        a[12] = relu_h(l3.x, r3.x);  a[13] = relu_h(l3.y, r3.y);
        a[14] = relu_h(l3.z, r3.z);  a[15] = relu_h(l3.w, r3.w);

        #pragma unroll
        for (int ct = 0; ct < 8; ++ct) {
            const v16h bm = w2v[(kt * 8 + ct) * 32 + lane];
            acc[ct] = __builtin_amdgcn_wmma_f32_16x16x32_f16(
                false, a, false, bm, (short)0, acc[ct], false, false);
        }
    }

    // ---- epilogue: edge[m] = sigmoid( sum_k relu(h2[m,k]+b2[k])*w3[k] + b3 )
    // C layout: lanes 0-15 hold rows 0-7 (vgpr r), lanes 16-31 rows 8-15,
    // column = ct*16 + (lane&15).
    float part[8] = {0.f,0.f,0.f,0.f,0.f,0.f,0.f,0.f};
    const int n16 = lane & 15;
    #pragma unroll
    for (int ct = 0; ct < 8; ++ct) {
        const int col = (ct << 4) + n16;
        const float w3v = W3[col];
        const float b2v = b2[col];
        #pragma unroll
        for (int r = 0; r < 8; ++r) {
            float v = acc[ct][r] + b2v;
            v = v > 0.f ? v : 0.f;
            part[r] += v * w3v;
        }
    }
    // butterfly reduce across the 16 lanes that share a row-set
    #pragma unroll
    for (int mask = 1; mask < 16; mask <<= 1) {
        #pragma unroll
        for (int r = 0; r < 8; ++r)
            part[r] += __shfl_xor(part[r], mask, 32);
    }

    const float b3v = b3[0];
    const int half = lane >> 4;
    const int r    = lane & 15;
    if (r < 8) {
        const int mrow = r + (half << 3);          // row 0-7 or 8-15
        const float z  = part[r] + b3v;
        const float e  = 1.f / (1.f + __expf(-z)); // sigmoid
        const int j = jbase + mrow;
        const size_t idx = (((size_t)b * NN + i) << 9) + j;
        E[idx] = base_adj[idx] * e + ((i == j) ? 1.f : 0.f);
    }
}

// ---------------------------------------------------------------------------
// Kernel 4: per-row softmax over E, agg = softmax @ x, out = agg@Wg + bg
// grid = B*N blocks, 256 threads
// ---------------------------------------------------------------------------
__global__ __launch_bounds__(256) void softmax_agg(const float* __restrict__ E,
                                                   const float* __restrict__ x,
                                                   const float* __restrict__ Wg,
                                                   const float* __restrict__ bg,
                                                   float* __restrict__ out) {
    __shared__ float sh[NN];     // softmax probs, then unused
    __shared__ float red[256];   // reductions / agg staging
    const int tid = threadIdx.x;
    const int row = blockIdx.x;          // b*N + i
    const int b   = row >> 9;

    const float* erow = E + ((size_t)row << 9);
    const float v0 = erow[tid];
    const float v1 = erow[tid + 256];

    red[tid] = fmaxf(v0, v1);
    __syncthreads();
    for (int s = 128; s > 0; s >>= 1) {
        if (tid < s) red[tid] = fmaxf(red[tid], red[tid + s]);
        __syncthreads();
    }
    const float mx = red[0];
    __syncthreads();

    const float e0 = __expf(v0 - mx);
    const float e1 = __expf(v1 - mx);
    red[tid] = e0 + e1;
    __syncthreads();
    for (int s = 128; s > 0; s >>= 1) {
        if (tid < s) red[tid] += red[tid + s];
        __syncthreads();
    }
    const float inv = 1.f / red[0];
    __syncthreads();

    sh[tid]       = e0 * inv;
    sh[tid + 256] = e1 * inv;
    __syncthreads();

    // agg[d=tid] = sum_j p[j] * x[b,j,d]
    const float* xb = x + ((size_t)b << 17);   // b*N*D
    float a = 0.f;
    #pragma unroll 8
    for (int j = 0; j < NN; ++j) a += sh[j] * xb[j * DD + tid];

    red[tid] = a;
    __syncthreads();

    // out[h=tid] = sum_d agg[d] * Wg[d,h] + bg[h]
    float o = bg[tid];
    #pragma unroll 8
    for (int d = 0; d < DD; ++d) o += red[d] * Wg[d * HH + tid];

    out[((size_t)row << 8) + tid] = o;
}

// ---------------------------------------------------------------------------
extern "C" void kernel_launch(void* const* d_in, const int* in_sizes, int n_in,
                              void* d_out, int out_size, void* d_ws, size_t ws_size,
                              hipStream_t stream) {
    const float* x        = (const float*)d_in[0];
    const float* base_adj = (const float*)d_in[1];
    const float* W1       = (const float*)d_in[2];
    const float* b1       = (const float*)d_in[3];
    const float* W2       = (const float*)d_in[4];
    const float* b2       = (const float*)d_in[5];
    const float* W3       = (const float*)d_in[6];
    const float* b3       = (const float*)d_in[7];
    const float* Wg       = (const float*)d_in[8];
    const float* bg       = (const float*)d_in[9];
    float* out = (float*)d_out;

    // workspace layout (floats): Lp[262144] | Rp[262144] | E[524288] | W2p(f16 32768)
    float*    Lp  = (float*)d_ws;
    float*    Rp  = Lp + (size_t)BB * NN * HH;
    float*    E   = Rp + (size_t)BB * NN * HH;
    _Float16* W2p = (_Float16*)(E + (size_t)BB * NN * NN);

    prep_lr   <<<BB * NN, 256, 0, stream>>>(x, W1, b1, Lp, Rp);
    pack_w2   <<<128,     256, 0, stream>>>(W2, W2p);
    // 32768 pair-tiles, 4 waves (tiles) per 128-thread block
    edge_mlp  <<<8192,    128, 0, stream>>>(Lp, Rp, W2p, b2, W3, b3, base_adj, E);
    softmax_agg<<<BB * NN, 256, 0, stream>>>(E, x, Wg, bg, out);
}
